// GraphAttentionLayer_31731218382937
// MI455X (gfx1250) — matrix operand order repaired and checked by reading, not compiled, not run on hardware
//
#include <hip/hip_runtime.h>

// ---------------------------------------------------------------------------
// GAT layer for MI455X (gfx1250, wave32, WMMA).
//  k1: Wh = h @ W             (bf16 WMMA, f32 accum) -> Wh f32 + Wh^T bf16
//  k2: f1 = Wh@a1, f2 = Wh@a2 (wave reductions)
//  k2b: gmax = max(f2)        (closed-form softmax stabilizer: rank-1 scores)
//  k3: fused mask + stable softmax + attention@Wh + ELU (single adj pass,
//      ping-pong P staging -> one barrier per 64-column tile)
// ---------------------------------------------------------------------------

typedef __attribute__((ext_vector_type(8)))  float  v8f;
typedef __attribute__((ext_vector_type(8)))  __bf16 v8bf;
typedef __attribute__((ext_vector_type(16))) __bf16 v16bf;

#define NEGV  -9.0e15f
#define ALPHA 0.2f
#define N_NODES 8192
#define F_IN    512
#define F_OUT   256

__device__ __forceinline__ unsigned short f2bf_bits(float f) {
  union { float f; unsigned u; } v; v.f = f;
  unsigned r = v.u + 0x7FFFu + ((v.u >> 16) & 1u);   // round-to-nearest-even
  return (unsigned short)(r >> 16);
}

__device__ __forceinline__ float leaky(float x) {    // slope 0<a<1: max(x, a*x)
  return fmaxf(x, ALPHA * x);
}

// A-fragment (16x32 bf16), contiguous K: lane l: row = l&15, kb = 8*(l>>4);
// elements 0..15 = K 2*kb .. 2*kb+15 (i.e. 16*(l>>4) + s).
__device__ __forceinline__ v16bf load_a_frag(const unsigned short* rowp, int kb) {
  v8bf lo = *(const v8bf*)(rowp + 2 * kb);
  v8bf hi = *(const v8bf*)(rowp + 2 * kb + 8);
  v16bf r;
#pragma unroll
  for (int i = 0; i < 8; ++i) { r[i] = lo[i]; r[i + 8] = hi[i]; }
  return r;
}

// ---------------------------------------------------------------------------
// Kernel 1: Wh = h(8192x512) @ W(512x256).  Block tile 128x64, K-step 32.
// 256 threads = 8 waves in 4(M) x 2(N); each wave -> 32x32 via 2x2 WMMA.
// ---------------------------------------------------------------------------
__global__ __launch_bounds__(256)
void gat_gemm1(const float* __restrict__ h, const float* __restrict__ W,
               float* __restrict__ Wh, unsigned short* __restrict__ WhT) {
  __shared__ __align__(64) unsigned short Al[128][32];   // row-major, 8 KB
  __shared__ __align__(64) unsigned short Btl[64][32];   // [col][k],  4 KB
  const int tid  = threadIdx.x;
  const int m0   = (blockIdx.x >> 2) * 128;
  const int n0   = (blockIdx.x & 3) * 64;
  const int wave = tid >> 5, lane = tid & 31;
  const int lh   = lane >> 4, ln = lane & 15;
  const int mw   = (wave >> 1) * 32, nw = (wave & 1) * 32;

  v8f acc[2][2] = {};
  for (int k0 = 0; k0 < F_IN; k0 += 32) {
#pragma unroll
    for (int it = 0; it < 4; ++it) {              // A tile: 128x32 f32 -> bf16
      int flat4 = tid + 256 * it;
      int row = flat4 >> 3, k4 = (flat4 & 7) << 2;
      float4 av = *(const float4*)(h + (long)(m0 + row) * F_IN + k0 + k4);
      Al[row][k4 + 0] = f2bf_bits(av.x);
      Al[row][k4 + 1] = f2bf_bits(av.y);
      Al[row][k4 + 2] = f2bf_bits(av.z);
      Al[row][k4 + 3] = f2bf_bits(av.w);
    }
#pragma unroll
    for (int it = 0; it < 2; ++it) {              // B tile: 32x64 -> [col][k]
      int flat4 = tid + 256 * it;
      int kr = flat4 >> 4, n4 = (flat4 & 15) << 2;
      float4 bv = *(const float4*)(W + (long)(k0 + kr) * F_OUT + n0 + n4);
      Btl[n4 + 0][kr] = f2bf_bits(bv.x);
      Btl[n4 + 1][kr] = f2bf_bits(bv.y);
      Btl[n4 + 2][kr] = f2bf_bits(bv.z);
      Btl[n4 + 3][kr] = f2bf_bits(bv.w);
    }
    __syncthreads();
    const int kb = lh * 8;
    v16bf a0 = load_a_frag(&Al[mw + ln][0], kb);
    v16bf a1 = load_a_frag(&Al[mw + 16 + ln][0], kb);
    v16bf b0 = *(const v16bf*)&Btl[nw + ln][16 * lh];
    v16bf b1 = *(const v16bf*)&Btl[nw + 16 + ln][16 * lh];
    acc[0][0] = __builtin_amdgcn_wmma_f32_16x16x32_bf16(false, a0, false, b0, (short)0, acc[0][0], false, false);
    acc[0][1] = __builtin_amdgcn_wmma_f32_16x16x32_bf16(false, a0, false, b1, (short)0, acc[0][1], false, false);
    acc[1][0] = __builtin_amdgcn_wmma_f32_16x16x32_bf16(false, a1, false, b0, (short)0, acc[1][0], false, false);
    acc[1][1] = __builtin_amdgcn_wmma_f32_16x16x32_bf16(false, a1, false, b1, (short)0, acc[1][1], false, false);
    __syncthreads();
  }
  // C layout: VGPR v -> row = v + 8*(lane>>4), col = lane&15
#pragma unroll
  for (int mi = 0; mi < 2; ++mi)
#pragma unroll
    for (int v = 0; v < 8; ++v) {
      int row = m0 + mw + 16 * mi + v + 8 * lh;
#pragma unroll
      for (int ni = 0; ni < 2; ++ni) {
        float val = acc[mi][ni][v];
        int col = n0 + nw + 16 * ni + ln;
        Wh[(long)row * F_OUT + col] = val;
        WhT[(long)col * N_NODES + row] = f2bf_bits(val);
      }
    }
}

// ---------------------------------------------------------------------------
// Kernel 2: f1[i] = Wh[i,:].a[0:256], f2[i] = Wh[i,:].a[256:512]
// ---------------------------------------------------------------------------
__global__ __launch_bounds__(256)
void gat_f12(const float* __restrict__ Wh, const float* __restrict__ a,
             float* __restrict__ f1, float* __restrict__ f2) {
  const int tid = threadIdx.x;
  const int row = blockIdx.x * 8 + (tid >> 5);
  const int lane = tid & 31;
  const float* wr = Wh + (long)row * F_OUT;
  float s1 = 0.f, s2 = 0.f;
#pragma unroll
  for (int t = 0; t < 8; ++t) {
    int k = lane + 32 * t;
    float v = wr[k];
    s1 += v * a[k];
    s2 += v * a[F_OUT + k];
  }
#pragma unroll
  for (int off = 16; off >= 1; off >>= 1) {
    s1 += __shfl_xor(s1, off, 32);
    s2 += __shfl_xor(s2, off, 32);
  }
  if (lane == 0) { f1[row] = s1; f2[row] = s2; }
}

// ---------------------------------------------------------------------------
// Kernel 2b: gmax = max_j f2[j]  (single block)
// ---------------------------------------------------------------------------
__global__ __launch_bounds__(256)
void gat_gmax(const float* __restrict__ f2, float* __restrict__ gmax) {
  __shared__ float red[8];
  const int tid = threadIdx.x;
  float m = -__builtin_inff();
  for (int idx = tid; idx < N_NODES; idx += 256) m = fmaxf(m, f2[idx]);
#pragma unroll
  for (int off = 16; off >= 1; off >>= 1) m = fmaxf(m, __shfl_xor(m, off, 32));
  if ((tid & 31) == 0) red[tid >> 5] = m;
  __syncthreads();
  if (tid == 0) {
    float r = red[0];
#pragma unroll
    for (int w = 1; w < 8; ++w) r = fmaxf(r, red[w]);
    gmax[0] = r;
  }
}

// ---------------------------------------------------------------------------
// Kernel 3: fused masked-softmax attention.  16 output rows per block,
// 256 threads = 8 waves; wave w owns output cols [32w, 32w+32).
// j-tiles of 64; stabilizer m_i = leaky(f1_i + max f2) is precomputed (no
// online max/rescale); ping-pong P buffers -> single barrier per tile; B
// fragments issued before the barrier so L2 latency overlaps the wait.
// ---------------------------------------------------------------------------
__global__ __launch_bounds__(256)
void gat_attn(const int* __restrict__ adj, const unsigned short* __restrict__ WhT,
              const float* __restrict__ f1g, const float* __restrict__ f2g,
              const float* __restrict__ gmaxp, float* __restrict__ out) {
  __shared__ __align__(16) float f2_s[N_NODES];              // 32 KB
  __shared__ __align__(64) unsigned short p_s[2][16][64];    // 4 KB ping-pong
  __shared__ float f1_s[16], s_s[16];

  const int tid  = threadIdx.x;
  const int i0   = blockIdx.x * 16;
  const int wave = tid >> 5, lane = tid & 31;
  const int lh   = lane >> 4, ln = lane & 15;

  {
    float4* d = (float4*)f2_s;
    const float4* s = (const float4*)f2g;
    for (int idx = tid; idx < N_NODES / 4; idx += 256) d[idx] = s[idx];
  }
  if (tid < 16) f1_s[tid] = f1g[i0 + tid];
  __syncthreads();

  const float gm = gmaxp[0];
  const int r = tid >> 4, c = tid & 15;           // half-wave = one row, 16 lanes x 4 cols
  const float f1r = f1_s[r];
  const float mr  = leaky(f1r + gm);              // per-row stabilizer, >= all e_ij
  const int* adjrow = adj + (long)(i0 + r) * N_NODES;
  const int c0a = wave * 32;
  const int c0b = wave * 32 + 16;
  const int kb  = lh * 8;
  const unsigned short* wt0 = WhT + (long)(c0a + ln) * N_NODES + 16 * lh;
  const unsigned short* wt1 = WhT + (long)(c0b + ln) * N_NODES + 16 * lh;

  v8f acc0 = {}, acc1 = {};
  float ssum = 0.0f;                              // deferred row-sum partial

  for (int t = 0; t < N_NODES / 64; ++t) {
    const int j0 = t * 64;
    const int buf = t & 1;
    // ---- scores + probabilities for 16x64 tile (4 adjacent cols/thread) ----
    int4   av = *(const int4*)(adjrow + j0 + 4 * c);
    float4 fv = *(const float4*)(f2_s + j0 + 4 * c);
    float e0 = av.x > 0 ? leaky(f1r + fv.x) : NEGV;
    float e1 = av.y > 0 ? leaky(f1r + fv.y) : NEGV;
    float e2 = av.z > 0 ? leaky(f1r + fv.z) : NEGV;
    float e3 = av.w > 0 ? leaky(f1r + fv.w) : NEGV;
    float p0 = __expf(e0 - mr);
    float p1 = __expf(e1 - mr);
    float p2 = __expf(e2 - mr);
    float p3 = __expf(e3 - mr);
    ssum += (p0 + p1) + (p2 + p3);
    union { unsigned short u[4]; uint2 v; } pk;
    pk.u[0] = f2bf_bits(p0); pk.u[1] = f2bf_bits(p1);
    pk.u[2] = f2bf_bits(p2); pk.u[3] = f2bf_bits(p3);
    *(uint2*)&p_s[buf][r][4 * c] = pk.v;
    // ---- B fragments (independent of P): overlap latency with barrier ----
    v16bf b00 = *(const v16bf*)(wt0 + j0);
    v16bf b01 = *(const v16bf*)(wt0 + j0 + 32);
    v16bf b10 = *(const v16bf*)(wt1 + j0);
    v16bf b11 = *(const v16bf*)(wt1 + j0 + 32);
    __syncthreads();
    // ---- WMMA accumulate: 2 K-steps of 32, 2 col-tiles ----
    const unsigned short* prow = &p_s[buf][ln][0];
    v16bf pf0 = load_a_frag(prow, kb);            // K  0..31
    v16bf pf1 = load_a_frag(prow + 32, kb);       // K 32..63
    acc0 = __builtin_amdgcn_wmma_f32_16x16x32_bf16(false, pf0, false, b00, (short)0, acc0, false, false);
    acc1 = __builtin_amdgcn_wmma_f32_16x16x32_bf16(false, pf0, false, b10, (short)0, acc1, false, false);
    acc0 = __builtin_amdgcn_wmma_f32_16x16x32_bf16(false, pf1, false, b01, (short)0, acc0, false, false);
    acc1 = __builtin_amdgcn_wmma_f32_16x16x32_bf16(false, pf1, false, b11, (short)0, acc1, false, false);
    // no trailing barrier: next iteration writes the other P buffer, and any
    // re-write of this buffer is ordered behind the next barrier + dscnt wait
  }

  // ---- final row-sum reduction over the 16 lanes of each half-wave ----
#pragma unroll
  for (int off = 8; off >= 1; off >>= 1) ssum += __shfl_xor(ssum, off, 32);
  if (c == 0) s_s[r] = ssum;
  __syncthreads();

  // ---- normalize + ELU ----
#pragma unroll
  for (int v = 0; v < 8; ++v) {
    int row = v + 8 * lh;
    float inv = 1.0f / s_s[row];
    float h0 = acc0[v] * inv;
    float h1 = acc1[v] * inv;
    out[(long)(i0 + row) * F_OUT + c0a + ln] = h0 > 0.0f ? h0 : __expf(h0) - 1.0f;
    out[(long)(i0 + row) * F_OUT + c0b + ln] = h1 > 0.0f ? h1 : __expf(h1) - 1.0f;
  }
}

// ---------------------------------------------------------------------------
extern "C" void kernel_launch(void* const* d_in, const int* in_sizes, int n_in,
                              void* d_out, int out_size, void* d_ws, size_t ws_size,
                              hipStream_t stream) {
  const float* h   = (const float*)d_in[0];   // [8192,512]
  const int*   adj = (const int*)d_in[1];     // [8192,8192]
  const float* W   = (const float*)d_in[2];   // [512,256]
  const float* a   = (const float*)d_in[3];   // [512,1]
  float* out = (float*)d_out;                 // [8192,256]

  char* ws = (char*)d_ws;
  float*          Wh   = (float*)ws;                          //  8 MB f32
  unsigned short* WhT  = (unsigned short*)(ws + 8388608);     //  4 MB bf16 [256][8192]
  float*          f1   = (float*)(ws + 12582912);             // 32 KB
  float*          f2   = (float*)(ws + 12615680);             // 32 KB
  float*          gmax = (float*)(ws + 12648448);             // 4 B

  gat_gemm1<<<dim3((N_NODES / 128) * (F_OUT / 64)), dim3(256), 0, stream>>>(h, W, Wh, WhT);
  gat_f12  <<<dim3(N_NODES / 8),                    dim3(256), 0, stream>>>(Wh, a, f1, f2);
  gat_gmax <<<dim3(1),                              dim3(256), 0, stream>>>(f2, gmax);
  gat_attn <<<dim3(N_NODES / 16),                   dim3(256), 0, stream>>>(adj, WhT, f1, f2, gmax, out);
}
